// CSIDiffusion_11527692222543
// MI455X (gfx1250) — matrix operand order repaired and checked by reading, not compiled, not run on hardware
//
#include <hip/hip_runtime.h>
#include <cmath>

typedef __attribute__((ext_vector_type(16))) _Float16 v16h;
typedef __attribute__((ext_vector_type(8)))  _Float16 v8h;
typedef __attribute__((ext_vector_type(4)))  _Float16 v4h;
typedef __attribute__((ext_vector_type(8)))  float    v8f;

#define NB 32
#define NH 128
#define NW 128
#define HW (NH*NW)
#define NCH 64
#define NEMB 128
#define NSTEPS 100
#define PH 130              // padded height (zero halo)
#define PW 130              // padded width

__device__ __forceinline__ float silu(float v) {
  // fast native path: v_exp_f32 + v_rcp_f32 (precision >> f16 storage)
  return v * __builtin_amdgcn_rcpf(1.0f + __expf(-v));
}

// ---------------- schedule: sqrt(alpha_bar[t]), sqrt(1-alpha_bar[t]) ----------------
__global__ void k_sched(const int* __restrict__ t, float* __restrict__ sqab,
                        float* __restrict__ sq1m) {
  int b = threadIdx.x;
  if (b >= NB) return;
  int tb = t[b];
  float prod = 1.0f;
  for (int i = 0; i <= tb; ++i) {
    float beta = 1e-4f + (0.2f - 1e-4f) * (float)i / (float)(NSTEPS - 1);
    prod *= (1.0f - beta);
  }
  sqab[b] = sqrtf(prod);
  sq1m[b] = sqrtf(1.0f - prod);
}

// ---------------- time embedding: emb[b][c] = silu(sincos(t)) @ temb_w.T + temb_b ----
__global__ void k_temb(const int* __restrict__ t, const float* __restrict__ tw,
                       const float* __restrict__ tbias, float* __restrict__ emb) {
  int idx = blockIdx.x * blockDim.x + threadIdx.x;
  if (idx >= NB * NCH) return;
  int b = idx / NCH, c = idx % NCH;
  float tf = (float)t[b];
  const float lg = 9.210340371976184f;   // log(10000)
  float acc = tbias[c];
  for (int k = 0; k < NEMB; ++k) {
    int j = k & 63;
    float fr = __expf(-lg * (float)j / 64.0f);
    float a = tf * fr;
    float e = (k < 64) ? __sinf(a) : __cosf(a);
    acc += silu(e) * tw[c * NEMB + k];
  }
  emb[idx] = acc;
}

// -------- forward noising + pilot conditioning -> x4 NHWC f16 [B][H][W][4] ----------
__global__ void k_pack(const float* __restrict__ x0, const float* __restrict__ mask,
                       const float* __restrict__ cond, const float* __restrict__ noise,
                       const float* __restrict__ sqab, const float* __restrict__ sq1m,
                       v4h* __restrict__ x4) {
  int i = blockIdx.x * blockDim.x + threadIdx.x;
  if (i >= NB * HW) return;
  int b = i / HW;
  float m = mask[i];
  float xt = sqab[b] * x0[i] + sq1m[b] * noise[i];
  xt = xt * (1.0f - m) + cond[i] * m;
  v4h v = { (_Float16)xt, (_Float16)m, (_Float16)cond[i], (_Float16)0.0f };
  x4[i] = v;
}

// -------- zero the halo ring of the padded h1/h2 NHWC tensors -----------------------
__global__ void k_ring(v8h* __restrict__ h1p, v8h* __restrict__ h2p) {
  int idx = blockIdx.x * blockDim.x + threadIdx.x;     // NB*516*8
  const int per = 516 * 8;
  if (idx >= NB * per) return;
  int b = idx / per, r8 = idx % per;
  int p = r8 >> 3, c8 = r8 & 7;
  int y, x;
  if (p < 130)      { y = -1;      x = p - 1;   }      // top row (incl. corners)
  else if (p < 260) { y = NH;      x = p - 131; }      // bottom row (incl. corners)
  else if (p < 388) { y = p - 260; x = -1;      }      // left col
  else              { y = p - 388; x = NW;      }      // right col
  size_t o = ((size_t)(b * PH + (y + 1)) * PW + (x + 1)) * 8 + c8;
  v8h z = {};
  h1p[o] = z;
  h2p[o] = z;
}

// -------- pack w1 into WMMA A-fragment layout; K = r*3+ci (27 padded to 32) ---------
__global__ void k_packA1(const float* __restrict__ w1, _Float16* __restrict__ Ap1) {
  int idx = blockIdx.x * blockDim.x + threadIdx.x;  // 4*32*16 = 2048
  if (idx >= 2048) return;
  int f    = idx & 15;
  int lane = (idx >> 4) & 31;
  int mt   = idx >> 9;
  int m  = mt * 16 + (lane & 15);
  int hb = (lane >> 4) * 8;
  int v = f >> 1, hp = f & 1;
  int kk = 2 * hb + 2 * v + hp;
  float val = 0.0f;
  if (kk < 27) {
    int r = kk / 3, ci = kk % 3;          // r = dy*3+dx
    val = w1[m * 27 + ci * 9 + r];        // w1[co][ci][dy][dx]
  }
  Ap1[idx] = (_Float16)val;
}

// -------- pack w2 into WMMA A-fragment layout; K = r*64+ci (576 = 18 chunks) --------
__global__ void k_packA(const float* __restrict__ w2, _Float16* __restrict__ Ap) {
  int idx = blockIdx.x * blockDim.x + threadIdx.x;  // 4*18*32*16 = 36864
  if (idx >= 4 * 18 * 512) return;
  int f    = idx & 15;
  int lane = (idx >> 4) & 31;
  int kc   = (idx >> 9) % 18;
  int mt   = idx / (18 * 512);
  int m  = mt * 16 + (lane & 15);
  int hb = (lane >> 4) * 8;
  int v = f >> 1, hp = f & 1;
  int kk = 2 * hb + 2 * v + hp;
  int kg = kc * 32 + kk;
  int r = kg / 64, ci = kg % 64;
  Ap[idx] = (_Float16)w2[(size_t)(m * 64 + ci) * 9 + r];
}

// ---------------- conv1 (3->64) WMMA + time-emb + silu -> h1p (padded NHWC) ---------
__global__ void k_conv1(const v4h* __restrict__ x4, const _Float16* __restrict__ Ap1,
                        const float* __restrict__ b1, const float* __restrict__ emb,
                        _Float16* __restrict__ h1p) {
  __shared__ __align__(32) _Float16 sp[64 * 32];   // [x][k]
  __shared__ float eb[64];
  int bid  = blockIdx.x;                 // (b*H + y)*2 + xblk
  int xblk = bid & 1;
  int row  = bid >> 1;
  int b = row / NH, y = row % NH;
  int x0 = xblk * 64;
  int tid = threadIdx.x;
  if (tid < 64) eb[tid] = b1[tid] + emb[b * 64 + tid];
  const _Float16* x4s = (const _Float16*)x4;
  for (int s = tid; s < 64 * 32; s += 128) {       // 16 iters
    int x = s & 63, k = s >> 6;
    _Float16 v = (_Float16)0.0f;
    if (k < 27) {
      int r = k / 3, ci = k % 3;
      int yy = y + r / 3 - 1, gx = x0 + x + (r % 3) - 1;
      if (yy >= 0 && yy < NH && gx >= 0 && gx < NW)
        v = x4s[(size_t)((b * NH + yy) * NW + gx) * 4 + ci];
    }
    sp[x * 32 + k] = v;
  }
  __syncthreads();

  int wave = tid >> 5, lane = tid & 31;
  int n0 = lane & 15, kb = (lane >> 4) << 4;
  v16h a = ((const v16h*)Ap1)[wave * 32 + lane];
  v8f acc[4] = {};
#pragma unroll
  for (int nt = 0; nt < 4; ++nt) {
    v16h bf = *(const v16h*)&sp[(nt * 16 + n0) * 32 + kb];
    acc[nt] = __builtin_amdgcn_wmma_f32_16x16x32_f16(
        false, a, false, bf, (short)0, acc[nt], false, false);
  }
  int mrow = (lane >> 4) * 8;
  int co8 = wave * 16 + mrow;
#pragma unroll
  for (int nt = 0; nt < 4; ++nt) {
    int gx = x0 + nt * 16 + n0;
    v8h hv;
#pragma unroll
    for (int v = 0; v < 8; ++v)
      hv[v] = (_Float16)silu(acc[nt][v] + eb[co8 + v]);
    *(v8h*)&h1p[((size_t)(b * PH + y + 1) * PW + (gx + 1)) * 64 + co8] = hv;
  }
}

// ---------------- conv2 (64->64) WMMA implicit GEMM, halo-padded NHWC ---------------
// B fragment = 16 consecutive channels at one pixel = one unconditional 32B load.
__global__ void k_conv2(const _Float16* __restrict__ h1p, const _Float16* __restrict__ Ap,
                        const float* __restrict__ b2, _Float16* __restrict__ h2p) {
  __shared__ float sb2[64];
  int bid  = blockIdx.x;                 // (b*H + y)*2 + xblk
  int xblk = bid & 1;
  int row  = bid >> 1;
  int b = row / NH, y = row % NH;
  int x0 = xblk * 64;
  int tid = threadIdx.x;
  if (tid < 64) sb2[tid] = b2[tid];
  __syncthreads();

  int wave = tid >> 5, lane = tid & 31;
  int n0 = lane & 15;
  int cib = (lane >> 4) << 4;            // 16*(lane/16)
  const v16h* Abase = (const v16h*)Ap + (size_t)wave * 18 * 32 + lane;

  v8f acc[4] = {};
  v16h a = Abase[0];
  for (int kc = 0; kc < 18; ++kc) {
    v16h an = a;
    if (kc < 17) an = Abase[(kc + 1) * 32];          // pipeline next A fragment
    int r = kc >> 1;                                 // dy*3+dx, fixed per chunk
    int dy = r / 3, dx = r % 3;
    int cb = 32 * (kc & 1) + cib;
    // padded row base: yy+1 = y+dy ; gx+1 = x0+nt*16+n0+dx
    const _Float16* base =
        h1p + ((size_t)(b * PH + y + dy) * PW) * 64 + cb + (size_t)(x0 + n0 + dx) * 64;
    v16h bf[4];
#pragma unroll
    for (int nt = 0; nt < 4; ++nt)
      bf[nt] = *(const v16h*)(base + (size_t)nt * 16 * 64);
#pragma unroll
    for (int nt = 0; nt < 4; ++nt)
      acc[nt] = __builtin_amdgcn_wmma_f32_16x16x32_f16(
          false, a, false, bf[nt], (short)0, acc[nt], false, false);
    a = an;
  }

  int mrow = (lane >> 4) * 8;
  int co8 = wave * 16 + mrow;
#pragma unroll
  for (int nt = 0; nt < 4; ++nt) {
    int gx = x0 + nt * 16 + n0;
    v8h hv;
#pragma unroll
    for (int v = 0; v < 8; ++v)
      hv[v] = (_Float16)silu(acc[nt][v] + sb2[co8 + v]);
    *(v8h*)&h2p[((size_t)(b * PH + y + 1) * PW + (gx + 1)) * 64 + co8] = hv;
  }
}

// ---------------- conv3 (64->1) + squared error + block reduction -------------------
__global__ void k_conv3(const _Float16* __restrict__ h2p, const float* __restrict__ w3,
                        const float* __restrict__ b3, const float* __restrict__ noise,
                        float* __restrict__ partials) {
  __shared__ __align__(16) _Float16 sp[3 * 130 * 64];  // [dy][x+1][ci]
  __shared__ float sw[576];
  __shared__ float red[128];
  int bid = blockIdx.x;                  // b*H + y
  int b = bid / NH, y = bid % NH;
  int tid = threadIdx.x;
  for (int i = tid; i < 576; i += 128) sw[i] = w3[i];
  // unconditional vector stage straight out of the padded tensor: 3120 v8h chunks
  for (int s = tid; s < 3 * 130 * 8; s += 128) {
    int c8 = (s & 7) * 8;
    int xx = (s >> 3) % 130;
    int dy = s / (130 * 8);
    v8h v = *(const v8h*)&h2p[((size_t)(b * PH + y + dy) * PW + xx) * 64 + c8];
    *(v8h*)&sp[(dy * 130 + xx) * 64 + c8] = v;
  }
  __syncthreads();
  int x = tid;
  float acc = b3[0];
#pragma unroll
  for (int dy = 0; dy < 3; ++dy)
#pragma unroll
    for (int dx = 0; dx < 3; ++dx) {
      const _Float16* rowp = &sp[(dy * 130 + (x + dx)) * 64];
      for (int c8 = 0; c8 < 64; c8 += 8) {
        v8h v = *(const v8h*)&rowp[c8];
#pragma unroll
        for (int j = 0; j < 8; ++j)
          acc += sw[(c8 + j) * 9 + dy * 3 + dx] * (float)v[j];
      }
    }
  float d = acc - noise[(size_t)b * HW + y * NW + x];
  red[tid] = d * d;
  __syncthreads();
  for (int off = 64; off > 0; off >>= 1) {
    if (tid < off) red[tid] += red[tid + off];
    __syncthreads();
  }
  if (tid == 0) partials[bid] = red[0];
}

__global__ void k_reduce(const float* __restrict__ partials, float* __restrict__ out) {
  __shared__ float red[256];
  int tid = threadIdx.x;
  float s = 0.0f;
  for (int i = tid; i < NB * NH; i += 256) s += partials[i];
  red[tid] = s;
  __syncthreads();
  for (int off = 128; off > 0; off >>= 1) {
    if (tid < off) red[tid] += red[tid + off];
    __syncthreads();
  }
  if (tid == 0) out[0] = red[0] / (float)(NB * HW);
}

// ------------------------------------------------------------------------------------
extern "C" void kernel_launch(void* const* d_in, const int* in_sizes, int n_in,
                              void* d_out, int out_size, void* d_ws, size_t ws_size,
                              hipStream_t stream) {
  const float* x0     = (const float*)d_in[0];
  const float* mask   = (const float*)d_in[1];
  const float* cond   = (const float*)d_in[2];
  const float* noise  = (const float*)d_in[3];
  const int*   t      = (const int*)  d_in[4];
  const float* w1     = (const float*)d_in[5];
  const float* b1     = (const float*)d_in[6];
  const float* tw     = (const float*)d_in[7];
  const float* tbias  = (const float*)d_in[8];
  const float* w2     = (const float*)d_in[9];
  const float* b2     = (const float*)d_in[10];
  const float* w3     = (const float*)d_in[11];
  const float* b3     = (const float*)d_in[12];
  float* out = (float*)d_out;

  char* ws = (char*)d_ws;
  size_t o = 0;
  auto alloc = [&](size_t bytes) { size_t r = o; o += (bytes + 255) & ~(size_t)255; return r; };
  float*    sqab     = (float*)   (ws + alloc(NB * sizeof(float)));
  float*    sq1m     = (float*)   (ws + alloc(NB * sizeof(float)));
  float*    emb      = (float*)   (ws + alloc(NB * NCH * sizeof(float)));
  _Float16* Ap       = (_Float16*)(ws + alloc(4 * 18 * 512 * sizeof(_Float16)));
  _Float16* Ap1      = (_Float16*)(ws + alloc(2048 * sizeof(_Float16)));
  v4h*      x4       = (v4h*)     (ws + alloc((size_t)NB * HW * sizeof(v4h)));
  _Float16* h1p      = (_Float16*)(ws + alloc((size_t)NB * PH * PW * NCH * sizeof(_Float16)));
  _Float16* h2p      = (_Float16*)(ws + alloc((size_t)NB * PH * PW * NCH * sizeof(_Float16)));
  float*    partials = (float*)   (ws + alloc(NB * NH * sizeof(float)));
  (void)ws_size; (void)in_sizes; (void)n_in; (void)out_size;

  k_sched  <<<1, 32, 0, stream>>>(t, sqab, sq1m);
  k_temb   <<<(NB * NCH + 255) / 256, 256, 0, stream>>>(t, tw, tbias, emb);
  k_pack   <<<(NB * HW) / 256, 256, 0, stream>>>(x0, mask, cond, noise, sqab, sq1m, x4);
  k_ring   <<<(NB * 516 * 8) / 256, 256, 0, stream>>>((v8h*)h1p, (v8h*)h2p);
  k_packA1 <<<2048 / 256, 256, 0, stream>>>(w1, Ap1);
  k_packA  <<<(4 * 18 * 512) / 256, 256, 0, stream>>>(w2, Ap);
  k_conv1  <<<NB * NH * 2, 128, 0, stream>>>(x4, Ap1, b1, emb, h1p);
  k_conv2  <<<NB * NH * 2, 128, 0, stream>>>(h1p, Ap, b2, h2p);
  k_conv3  <<<NB * NH, 128, 0, stream>>>(h2p, w3, b3, noise, partials);
  k_reduce <<<1, 256, 0, stream>>>(partials, out);
}
